// MoE_29394756174449
// MI455X (gfx1250) — matrix operand order repaired and checked by reading, not compiled, not run on hardware
//
#include <hip/hip_runtime.h>
#include <hip/hip_bf16.h>
#include <stdint.h>

// ---------------- problem constants (match reference) ----------------
#define NE     8
#define TOPK   2
#define DM     1024
#define DF     2048
#define TTOK   (4 * 2048)          // B*S = 8192 tokens
#define SLOTS  (TTOK * TOPK)       // 16384 routed rows total

// ---------------- GEMM tiling ----------------
#define TM   128
#define TN   128
#define TKS  64                    // staged K per round (2 WMMA k-chunks)
#define LDSS 72                    // LDS row stride in bf16 elems (144B, 16B aligned, skewed)

typedef __attribute__((ext_vector_type(16))) __bf16 v16bf;
typedef __attribute__((ext_vector_type(8)))  float  v8f;
typedef __attribute__((ext_vector_type(4)))  int    v4i;

#define AS1 __attribute__((address_space(1)))
#define AS3 __attribute__((address_space(3)))

// ---- CDNA5 async global->LDS copy (ASYNCcnt path), with safe fallback ----
#if __has_builtin(__builtin_amdgcn_global_load_async_to_lds_b128)
#define ASYNC_EN 1
#else
#define ASYNC_EN 0
#endif

__device__ __forceinline__ void cp16_g2l(const unsigned short* g, unsigned short* l) {
#if ASYNC_EN
    __builtin_amdgcn_global_load_async_to_lds_b128(
        (AS1 v4i*)g, (AS3 v4i*)l, 0, 0);
#else
    *(uint4*)l = *(const uint4*)g;
#endif
}

__device__ __forceinline__ void async_wait0() {
#if ASYNC_EN
#if __has_builtin(__builtin_amdgcn_s_wait_asynccnt)
    __builtin_amdgcn_s_wait_asynccnt(0);
#else
    asm volatile("s_wait_asynccnt 0x0" ::: "memory");
#endif
#endif
}

__device__ __forceinline__ unsigned short f32_to_bf16_rne(float f) {
    union { float f; uint32_t u; } v; v.f = f;
    uint32_t b = v.u;
    b += 0x7FFFu + ((b >> 16) & 1u);
    return (unsigned short)(b >> 16);
}

__device__ __forceinline__ float gelu_tanh(float x) {
    // jax.nn.gelu default (approximate=True)
    float x3 = x * x * x;
    return 0.5f * x * (1.0f + tanhf(0.7978845608028654f * (x + 0.044715f * x3)));
}

// =====================================================================
// Kernel 1: gating (one wave32 per token) + x -> bf16 conversion +
//           atomic scatter of (token, slot) into per-expert lists.
// =====================================================================
__global__ void __launch_bounds__(256)
moe_gate_kernel(const float* __restrict__ x,
                const float* __restrict__ Wg,
                const float* __restrict__ bg,
                unsigned short* __restrict__ x_bf,
                float* __restrict__ wts,
                int* __restrict__ tok_list,
                int* __restrict__ cnt) {
    const int wid  = threadIdx.x >> 5;
    const int lane = threadIdx.x & 31;
    const int t    = blockIdx.x * 8 + wid;

    float acc[NE];
#pragma unroll
    for (int e = 0; e < NE; ++e) acc[e] = 0.0f;

    const float* xr = x + (size_t)t * DM;
#pragma unroll 4
    for (int d = lane; d < DM; d += 32) {
        float xv = xr[d];
        x_bf[(size_t)t * DM + d] = f32_to_bf16_rne(xv);
        const float4* wrow = (const float4*)(Wg + (size_t)d * NE);
        float4 w0 = wrow[0], w1 = wrow[1];
        acc[0] += xv * w0.x; acc[1] += xv * w0.y;
        acc[2] += xv * w0.z; acc[3] += xv * w0.w;
        acc[4] += xv * w1.x; acc[5] += xv * w1.y;
        acc[6] += xv * w1.z; acc[7] += xv * w1.w;
    }
#pragma unroll
    for (int e = 0; e < NE; ++e)
#pragma unroll
        for (int off = 16; off > 0; off >>= 1)
            acc[e] += __shfl_xor(acc[e], off, 32);

    if (lane == 0) {
        float a[NE];
#pragma unroll
        for (int e = 0; e < NE; ++e) a[e] = acc[e] + bg[e];
        int e0 = 0;
#pragma unroll
        for (int e = 1; e < NE; ++e) if (a[e] > a[e0]) e0 = e;   // first max on tie
        int e1 = (e0 == 0) ? 1 : 0;
#pragma unroll
        for (int e = 0; e < NE; ++e) if (e != e0 && a[e] > a[e1]) e1 = e;
        float p  = __expf(a[e1] - a[e0]);
        float w0 = 1.0f / (1.0f + p);
        float w1 = p * w0;
        wts[2 * t + 0] = w0;
        wts[2 * t + 1] = w1;
        int p0 = atomicAdd(&cnt[e0], 1);
        tok_list[e0 * TTOK + p0] = (t << 1);
        int p1 = atomicAdd(&cnt[e1], 1);
        tok_list[e1 * TTOK + p1] = (t << 1) | 1;
    }
}

// =====================================================================
// Kernel 2: exclusive scan over 8 expert counts (trivial)
// =====================================================================
__global__ void moe_scan_kernel(const int* __restrict__ cnt, int* __restrict__ base) {
    if (threadIdx.x == 0 && blockIdx.x == 0) {
        int run = 0;
        for (int e = 0; e < NE; ++e) { base[e] = run; run += cnt[e]; }
    }
}

// =====================================================================
// Kernel 3: weight transpose + f32->bf16 (once per launch)
//   src: [E][R][C] f32  ->  dst: [E][C][R] bf16
// =====================================================================
__global__ void __launch_bounds__(256)
moe_transpose_kernel(const float* __restrict__ src, unsigned short* __restrict__ dst,
                     int R, int C) {
    __shared__ float tile[32][33];
    const int e  = blockIdx.z;
    const float*    s = src + (size_t)e * R * C;
    unsigned short* d = dst + (size_t)e * R * C;
    const int c0 = blockIdx.x * 32, r0 = blockIdx.y * 32;
    const int tx = threadIdx.x & 31, ty = threadIdx.x >> 5;   // 32 x 8
#pragma unroll
    for (int i = 0; i < 32; i += 8)
        tile[ty + i][tx] = s[(size_t)(r0 + ty + i) * C + c0 + tx];
    __syncthreads();
#pragma unroll
    for (int i = 0; i < 32; i += 8)
        d[(size_t)(c0 + ty + i) * R + r0 + tx] = f32_to_bf16_rne(tile[tx][ty + i]);
}

// =====================================================================
// Kernel 4: per-expert gather-GEMM  h = gelu(x_sel @ W1[e] + b1[e])
//   B is pre-transposed bf16: W1T [E][DF][DM]
//   grid = (DF/TN, TTOK/TM, NE), block = 256 (8 waves, 4x2 wave grid)
//   Double-buffered LDS; async copies overlap WMMA compute.
// =====================================================================
__global__ void __launch_bounds__(256)
moe_ffn1_kernel(const unsigned short* __restrict__ x_bf,
                const unsigned short* __restrict__ W1T,
                const float* __restrict__ b1,
                const int* __restrict__ tok_list,
                const int* __restrict__ cnt,
                const int* __restrict__ base,
                const unsigned short* __restrict__ zrow,
                unsigned short* __restrict__ h_bf) {
    __shared__ __align__(16) unsigned short As[2 * TM * LDSS];
    __shared__ __align__(16) unsigned short Bs[2 * TN * LDSS];

    const int e    = blockIdx.z;
    const int n0   = blockIdx.x * TN;
    const int m0   = blockIdx.y * TM;
    const int rows = cnt[e];
    if (m0 >= rows) return;               // over-provisioned grid: early exit

    const unsigned short* W1Te = W1T + (size_t)e * DM * DF;
    const int tid  = threadIdx.x;
    const int wid  = tid >> 5, lane = tid & 31;
    const int wm   = wid >> 1, wn = wid & 1;          // 4x2 wave grid (M x N)
    const int lh   = lane >> 4, r16 = lane & 15;

    auto stage = [&](int buf, int k0) {
        unsigned short* Ab = As + buf * (TM * LDSS);
        unsigned short* Bb = Bs + buf * (TN * LDSS);
#pragma unroll
        for (int i = 0; i < 4; ++i) {                 // A: gathered token rows
            int c  = tid + i * 256;                   // 0..1023
            int rr = c >> 3, cj = c & 7;
            const unsigned short* src;
            if (m0 + rr < rows) {
                int t = tok_list[e * TTOK + m0 + rr] >> 1;
                src = x_bf + (size_t)t * DM + k0 + cj * 8;
            } else {
                src = zrow + cj * 8;                  // zero page
            }
            cp16_g2l(src, Ab + rr * LDSS + cj * 8);
        }
#pragma unroll
        for (int i = 0; i < 4; ++i) {                 // B: dense pre-transposed
            int c  = tid + i * 256;
            int nn = c >> 3, cj = c & 7;
            cp16_g2l(W1Te + (size_t)(n0 + nn) * DM + k0 + cj * 8,
                     Bb + nn * LDSS + cj * 8);
        }
    };

    v8f acc[2][4] = {};                               // 32x64 per wave
    const int NT = DM / TKS;

    stage(0, 0);
    for (int kt = 0; kt < NT; ++kt) {
        async_wait0();
        __syncthreads();          // cur copies visible; prev reads of nxt done
        if (kt + 1 < NT) stage((kt + 1) & 1, (kt + 1) * TKS);

        const unsigned short* Ab = As + (kt & 1) * (TM * LDSS);
        const unsigned short* Bb = Bs + (kt & 1) * (TN * LDSS);
#pragma unroll
        for (int kk = 0; kk < 2; ++kk) {
            v16bf af[2], bfv[4];
#pragma unroll
            for (int mt = 0; mt < 2; ++mt) {
                const unsigned short* ap =
                    Ab + (wm * 32 + mt * 16 + r16) * LDSS + kk * 32;
                uint4* dst = (uint4*)&af[mt];
                dst[0] = *(const uint4*)(ap + lh * 8);        // K = 8*half+0..7
                dst[1] = *(const uint4*)(ap + 16 + lh * 8);   // K = 16+8*half+0..7
            }
#pragma unroll
            for (int nt = 0; nt < 4; ++nt) {
                const unsigned short* bp =
                    Bb + (wn * 64 + nt * 16 + r16) * LDSS + kk * 32;
                uint4* dst = (uint4*)&bfv[nt];
                dst[0] = *(const uint4*)(bp + lh * 8);        // K = 8*half+0..7
                dst[1] = *(const uint4*)(bp + 16 + lh * 8);   // K = 16+8*half+0..7
            }
#pragma unroll
            for (int mt = 0; mt < 2; ++mt)
#pragma unroll
                for (int nt = 0; nt < 4; ++nt)
                    acc[mt][nt] = __builtin_amdgcn_wmma_f32_16x16x32_bf16(
                        false, af[mt], false, bfv[nt],
                        (short)0, acc[mt][nt], false, false);
        }
    }

    // ---- epilogue: bias + gelu, store compacted bf16 rows ----
    const int hb = base[e];
#pragma unroll
    for (int nt = 0; nt < 4; ++nt) {
        const int nn = n0 + wn * 64 + nt * 16 + r16;
        const float bb = b1[e * DF + nn];
#pragma unroll
        for (int mt = 0; mt < 2; ++mt) {
#pragma unroll
            for (int i = 0; i < 8; ++i) {
                int row = wm * 32 + mt * 16 + lh * 8 + i;     // D-layout: m = i + 8*half
                if (m0 + row < rows) {
                    float v = gelu_tanh(acc[mt][nt][i] + bb);
                    h_bf[(size_t)(hb + m0 + row) * DF + nn] = f32_to_bf16_rne(v);
                }
            }
        }
    }
}

// =====================================================================
// Kernel 5: per-expert GEMM  y = h @ W2[e] + b2[e], scaled combine
//   B is pre-transposed bf16: W2T [E][DM][DF]
//   grid = (DM/TN, TTOK/TM, NE); double-buffered LDS
// =====================================================================
__global__ void __launch_bounds__(256)
moe_ffn2_kernel(const unsigned short* __restrict__ h_bf,
                const unsigned short* __restrict__ W2T,
                const float* __restrict__ b2,
                const int* __restrict__ tok_list,
                const int* __restrict__ cnt,
                const int* __restrict__ base,
                const float* __restrict__ wts,
                const unsigned short* __restrict__ zrow,
                float* __restrict__ out) {
    __shared__ __align__(16) unsigned short As[2 * TM * LDSS];
    __shared__ __align__(16) unsigned short Bs[2 * TN * LDSS];

    const int e    = blockIdx.z;
    const int n0   = blockIdx.x * TN;
    const int m0   = blockIdx.y * TM;
    const int rows = cnt[e];
    if (m0 >= rows) return;

    const unsigned short* W2Te = W2T + (size_t)e * DF * DM;
    const int hb   = base[e];
    const int tid  = threadIdx.x;
    const int wid  = tid >> 5, lane = tid & 31;
    const int wm   = wid >> 1, wn = wid & 1;
    const int lh   = lane >> 4, r16 = lane & 15;

    auto stage = [&](int buf, int k0) {
        unsigned short* Ab = As + buf * (TM * LDSS);
        unsigned short* Bb = Bs + buf * (TN * LDSS);
#pragma unroll
        for (int i = 0; i < 4; ++i) {                 // A: dense compacted h rows
            int c  = tid + i * 256;
            int rr = c >> 3, cj = c & 7;
            const unsigned short* src =
                (m0 + rr < rows)
                    ? h_bf + (size_t)(hb + m0 + rr) * DF + k0 + cj * 8
                    : zrow + cj * 8;
            cp16_g2l(src, Ab + rr * LDSS + cj * 8);
        }
#pragma unroll
        for (int i = 0; i < 4; ++i) {                 // B: dense pre-transposed
            int c  = tid + i * 256;
            int nn = c >> 3, cj = c & 7;
            cp16_g2l(W2Te + (size_t)(n0 + nn) * DF + k0 + cj * 8,
                     Bb + nn * LDSS + cj * 8);
        }
    };

    v8f acc[2][4] = {};
    const int NT = DF / TKS;

    stage(0, 0);
    for (int kt = 0; kt < NT; ++kt) {
        async_wait0();
        __syncthreads();
        if (kt + 1 < NT) stage((kt + 1) & 1, (kt + 1) * TKS);

        const unsigned short* Ab = As + (kt & 1) * (TM * LDSS);
        const unsigned short* Bb = Bs + (kt & 1) * (TN * LDSS);
#pragma unroll
        for (int kk = 0; kk < 2; ++kk) {
            v16bf af[2], bfv[4];
#pragma unroll
            for (int mt = 0; mt < 2; ++mt) {
                const unsigned short* ap =
                    Ab + (wm * 32 + mt * 16 + r16) * LDSS + kk * 32;
                uint4* dst = (uint4*)&af[mt];
                dst[0] = *(const uint4*)(ap + lh * 8);
                dst[1] = *(const uint4*)(ap + 16 + lh * 8);
            }
#pragma unroll
            for (int nt = 0; nt < 4; ++nt) {
                const unsigned short* bp =
                    Bb + (wn * 64 + nt * 16 + r16) * LDSS + kk * 32;
                uint4* dst = (uint4*)&bfv[nt];
                dst[0] = *(const uint4*)(bp + lh * 8);
                dst[1] = *(const uint4*)(bp + 16 + lh * 8);
            }
#pragma unroll
            for (int mt = 0; mt < 2; ++mt)
#pragma unroll
                for (int nt = 0; nt < 4; ++nt)
                    acc[mt][nt] = __builtin_amdgcn_wmma_f32_16x16x32_bf16(
                        false, af[mt], false, bfv[nt],
                        (short)0, acc[mt][nt], false, false);
        }
    }

    // ---- epilogue: bias, scale by routing weight, atomic combine ----
#pragma unroll
    for (int mt = 0; mt < 2; ++mt) {
        int   tarr[8];  float warr[8];  bool ok[8];
#pragma unroll
        for (int i = 0; i < 8; ++i) {
            int row = m0 + wm * 32 + mt * 16 + lh * 8 + i;
            ok[i] = (row < rows);
            if (ok[i]) {
                int entry = tok_list[e * TTOK + row];
                tarr[i] = entry >> 1;
                warr[i] = wts[2 * tarr[i] + (entry & 1)];
            } else { tarr[i] = 0; warr[i] = 0.0f; }
        }
#pragma unroll
        for (int nt = 0; nt < 4; ++nt) {
            const int nn = n0 + wn * 64 + nt * 16 + r16;
            const float bb = b2[e * DM + nn];
#pragma unroll
            for (int i = 0; i < 8; ++i) {
                if (ok[i]) {
                    // exactly two commutative f32 adds per out element -> deterministic
                    unsafeAtomicAdd(out + (size_t)tarr[i] * DM + nn,
                                    (acc[mt][nt][i] + bb) * warr[i]);
                }
            }
        }
    }
}

// =====================================================================
// Host launch
// =====================================================================
static inline size_t alignup(size_t v, size_t a) { return (v + a - 1) / a * a; }

extern "C" void kernel_launch(void* const* d_in, const int* in_sizes, int n_in,
                              void* d_out, int out_size, void* d_ws, size_t ws_size,
                              hipStream_t stream) {
    const float* x  = (const float*)d_in[0];   // [T, DM]
    const float* Wg = (const float*)d_in[1];   // [DM, NE]
    const float* bg = (const float*)d_in[2];   // [NE]
    const float* W1 = (const float*)d_in[3];   // [NE, DM, DF]
    const float* b1 = (const float*)d_in[4];   // [NE, DF]
    const float* W2 = (const float*)d_in[5];   // [NE, DF, DM]
    const float* b2 = (const float*)d_in[6];   // [NE, DM]
    float* out = (float*)d_out;                // [T, DM]

    // workspace carve-up
    char* ws = (char*)d_ws;
    size_t off = 0;
    unsigned short* x_bf = (unsigned short*)(ws + off);
    off = alignup(off + (size_t)TTOK * DM * sizeof(unsigned short), 256);
    unsigned short* h_bf = (unsigned short*)(ws + off);
    off = alignup(off + (size_t)SLOTS * DF * sizeof(unsigned short), 256);
    unsigned short* W1T = (unsigned short*)(ws + off);
    off = alignup(off + (size_t)NE * DM * DF * sizeof(unsigned short), 256);
    unsigned short* W2T = (unsigned short*)(ws + off);
    off = alignup(off + (size_t)NE * DF * DM * sizeof(unsigned short), 256);
    int* tok_list = (int*)(ws + off);
    off = alignup(off + (size_t)NE * TTOK * sizeof(int), 256);
    float* wts = (float*)(ws + off);
    off = alignup(off + (size_t)SLOTS * sizeof(float), 256);
    int* cnt = (int*)(ws + off);
    off = alignup(off + NE * sizeof(int), 256);
    int* base = (int*)(ws + off);
    off = alignup(off + NE * sizeof(int), 256);
    unsigned short* zrow = (unsigned short*)(ws + off);   // 256B zero page
    off = alignup(off + 256, 256);

    (void)hipMemsetAsync(out, 0, (size_t)TTOK * DM * sizeof(float), stream);
    (void)hipMemsetAsync(cnt, 0, NE * sizeof(int), stream);
    (void)hipMemsetAsync(zrow, 0, 256, stream);

    moe_gate_kernel<<<TTOK / 8, 256, 0, stream>>>(x, Wg, bg, x_bf, wts, tok_list, cnt);
    moe_scan_kernel<<<1, 32, 0, stream>>>(cnt, base);
    // W1 [E][DM][DF] -> W1T [E][DF][DM];  W2 [E][DF][DM] -> W2T [E][DM][DF]
    moe_transpose_kernel<<<dim3(DF / 32, DM / 32, NE), 256, 0, stream>>>(W1, W1T, DM, DF);
    moe_transpose_kernel<<<dim3(DM / 32, DF / 32, NE), 256, 0, stream>>>(W2, W2T, DF, DM);
    moe_ffn1_kernel<<<dim3(DF / TN, TTOK / TM, NE), 256, 0, stream>>>(
        x_bf, W1T, b1, tok_list, cnt, base, zrow, h_bf);
    moe_ffn2_kernel<<<dim3(DM / TN, TTOK / TM, NE), 256, 0, stream>>>(
        h_bf, W2T, b2, tok_list, cnt, base, wts, zrow, out);
}
